// MultiHeadAdaIn_27736898798336
// MI455X (gfx1250) — matrix operand order repaired and checked by reading, not compiled, not run on hardware
//
#include <hip/hip_runtime.h>
#include <hip/hip_bf16.h>

typedef __attribute__((ext_vector_type(16))) __bf16 v16bf;
typedef __attribute__((ext_vector_type(8)))  float  v8f;

#define BN    4
#define NPTS  16384
#define INCH  256
#define LSTYLE 256
#define NH    8
#define CCH   32
#define KVC   280     // NH*(CCH+3)
#define HC    256     // NH*CCH
#define GG    32
#define G3    32768

struct bf2 { __bf16 x, y; };

__device__ __forceinline__ int a_kmap(int e, int h) {
  // 16-bit A fragment: element e of v16bf -> K index (contiguous: k = 16h + e)
  return e + 16 * h;
}

// ---------------------------------------------------------------- zero
__global__ void k_zero(float* __restrict__ p, long long count) {
  long long i = (long long)blockIdx.x * blockDim.x + threadIdx.x;
  if (i < count) p[i] = 0.0f;
}

// ---------------------------------------------------------------- x -> bf16, transposed to [b][n][k]
__global__ void k_xpose(const float* __restrict__ x, __bf16* __restrict__ xT) {
  __shared__ float tile[32][65];
  const int b = blockIdx.z;
  const int k0 = blockIdx.y * 32;
  const int n0 = blockIdx.x * 64;
  for (int i = threadIdx.x; i < 32 * 64; i += 256) {
    int kl = i >> 6, nl = i & 63;
    tile[kl][nl] = x[((size_t)b * INCH + k0 + kl) * NPTS + n0 + nl];
  }
  __syncthreads();
  for (int i = threadIdx.x; i < 64 * 16; i += 256) {
    int nl = i >> 4, kp = i & 15;
    bf2 v;
    v.x = (__bf16)tile[2 * kp][nl];
    v.y = (__bf16)tile[2 * kp + 1][nl];
    *(bf2*)(xT + ((size_t)b * NPTS + n0 + nl) * INCH + k0 + 2 * kp) = v;
  }
}

// ---------------------------------------------------------------- grid f32 -> bf16 (channel-minor kept)
__global__ void k_cvtg(const float* __restrict__ g, __bf16* __restrict__ gb, long long npairs) {
  long long i = (long long)blockIdx.x * blockDim.x + threadIdx.x;
  if (i < npairs) {
    float2 v = ((const float2*)g)[i];
    bf2 o; o.x = (__bf16)v.x; o.y = (__bf16)v.y;
    ((bf2*)gb)[i] = o;
  }
}

// ---------------------------------------------------------------- kv GEMM (WMMA bf16)
// kv[b,o,n] = sum_i kv_w[o,i] * xT[b,n,i].  Block: 128 thr = 4 waves, wave: 16M x 64N.
__global__ void k_kv_gemm(const __bf16* __restrict__ xT, const float* __restrict__ kv_w,
                          float* __restrict__ kv) {
  __shared__ __attribute__((aligned(32))) __bf16 lAf[32 * 8 * 16]; // [lane][kchunk][e]
  const int b  = blockIdx.z;
  const int mt = blockIdx.y;
  const int tid = threadIdx.x;
  // stage A pre-swizzled into per-lane fragment order
  for (int i = tid; i < 4096; i += 128) {
    int e = i & 15, kc = (i >> 4) & 7, l = i >> 7;
    int m = mt * 16 + (l & 15);
    int k = kc * 32 + a_kmap(e, l >> 4);
    lAf[i] = (m < KVC) ? (__bf16)kv_w[m * INCH + k] : (__bf16)0.0f;
  }
  __syncthreads();
  const int wave = tid >> 5, lane = tid & 31;
  const int h = lane >> 4, lm = lane & 15;
  const int nbase = blockIdx.x * 256 + wave * 64;
  v8f acc[4];
  for (int t = 0; t < 4; ++t) acc[t] = {};
  const __bf16* xb = xT + (size_t)b * NPTS * INCH;
  for (int kc = 0; kc < 8; ++kc) {
    v16bf a = *(const v16bf*)&lAf[(lane * 8 + kc) * 16];
#pragma unroll
    for (int t = 0; t < 4; ++t) {
      int n = nbase + t * 16 + lm;
      v16bf bf = *(const v16bf*)(xb + (size_t)n * INCH + kc * 32 + 16 * h);
      acc[t] = __builtin_amdgcn_wmma_f32_16x16x32_bf16(
          false, a, false, bf, (short)0, acc[t], false, false);
    }
  }
#pragma unroll
  for (int t = 0; t < 4; ++t) {
    int n = nbase + t * 16 + lm;
#pragma unroll
    for (int r = 0; r < 8; ++r) {
      int o = mt * 16 + r + 8 * h;
      if (o < KVC) kv[((size_t)b * KVC + o) * NPTS + n] = acc[t][r];
    }
  }
}

// ---------------------------------------------------------------- per-channel instance-norm stats
__global__ void k_stats(const float* __restrict__ src, int CH,
                        float* __restrict__ mean, float* __restrict__ istd) {
  const int ch = blockIdx.x, b = blockIdx.y;
  const float* p = src + ((size_t)b * CH + ch) * NPTS;
  float s = 0.f, s2 = 0.f;
  for (int i = threadIdx.x; i < NPTS; i += 256) { float v = p[i]; s += v; s2 += v * v; }
  __shared__ float sh[256], sh2[256];
  sh[threadIdx.x] = s; sh2[threadIdx.x] = s2;
  __syncthreads();
  for (int off = 128; off > 0; off >>= 1) {
    if (threadIdx.x < off) { sh[threadIdx.x] += sh[threadIdx.x + off]; sh2[threadIdx.x] += sh2[threadIdx.x + off]; }
    __syncthreads();
  }
  if (threadIdx.x == 0) {
    float m = sh[0] / (float)NPTS;
    float v = sh2[0] / (float)NPTS - m * m;
    mean[b * CH + ch] = m;
    istd[b * CH + ch] = rsqrtf(v + 1e-5f);
  }
}

// ---------------------------------------------------------------- style affine: sb = style @ W.T + b
__global__ void k_affine(const float* __restrict__ style, const float* __restrict__ W,
                         const float* __restrict__ bias, float* __restrict__ out, int OUT) {
  const int o = blockIdx.x, b = blockIdx.y;
  float v = style[b * LSTYLE + threadIdx.x] * W[o * LSTYLE + threadIdx.x];
  __shared__ float sh[256];
  sh[threadIdx.x] = v; __syncthreads();
  for (int off = 128; off > 0; off >>= 1) {
    if (threadIdx.x < off) sh[threadIdx.x] += sh[threadIdx.x + off];
    __syncthreads();
  }
  if (threadIdx.x == 0) out[b * OUT + o] = sh[0] + bias[o];
}

// ---------------------------------------------------------------- adain keys/values + corners
__global__ void k_prep(const float* __restrict__ kv, const float* __restrict__ mean,
                       const float* __restrict__ istd, const float* __restrict__ sbk,
                       const float* __restrict__ sbv, const float* __restrict__ orig,
                       const float* __restrict__ scale_p, const float* __restrict__ R,
                       const float* __restrict__ t, int* __restrict__ idxb,
                       float* __restrict__ wb, float* __restrict__ vals) {
  const int b = blockIdx.y;
  const int n = blockIdx.x * 256 + threadIdx.x;
  const float scale = scale_p[0];
  float op[3];
  for (int d = 0; d < 3; ++d) op[d] = orig[((size_t)b * 3 + d) * NPTS + n];
  const float* kvb = kv + (size_t)b * KVC * NPTS;
  const float* mb = mean + b * KVC; const float* ib = istd + b * KVC;
  for (int hh = 0; hh < NH; ++hh) {
    float p[3];
    for (int d = 0; d < 3; ++d) {
      int ch = hh * 3 + d;
      float xn = (kvb[(size_t)ch * NPTS + n] - mb[ch]) * ib[ch];
      float g = 1.f + sbk[b * 48 + ch], bt = sbk[b * 48 + 24 + ch];
      p[d] = op[d] + scale * (g * xn + bt);
    }
    float pos[3];
    for (int i = 0; i < 3; ++i) {
      float key = R[hh * 9 + i * 3 + 0] * p[0] + R[hh * 9 + i * 3 + 1] * p[1] +
                  R[hh * 9 + i * 3 + 2] * p[2] + t[hh * 3 + i];
      float q = (tanhf(key) + 1.f) * (0.5f * (GG - 1));
      pos[i] = fminf(fmaxf(q, 0.f), (float)(GG - 1) - 1e-4f);
    }
    int i0[3]; float f[3];
    for (int i = 0; i < 3; ++i) { float fl = floorf(pos[i]); i0[i] = (int)fl; f[i] = pos[i] - fl; }
    const int bh = b * NH + hh;
#pragma unroll
    for (int o = 0; o < 8; ++o) {
      int bx = (o >> 2) & 1, by = (o >> 1) & 1, bz = o & 1;
      int idx = ((i0[0] + bx) * GG + (i0[1] + by)) * GG + (i0[2] + bz);
      float w = (bx ? f[0] : 1.f - f[0]) * (by ? f[1] : 1.f - f[1]) * (bz ? f[2] : 1.f - f[2]);
      idxb[((size_t)bh * 8 + o) * NPTS + n] = idx;
      wb[((size_t)bh * 8 + o) * NPTS + n] = w;
    }
    float vloc[CCH];
    for (int c = 0; c < CCH; ++c) {
      int ch = NH * 3 + hh * CCH + c;
      float xn = (kvb[(size_t)ch * NPTS + n] - mb[ch]) * ib[ch];
      float g = 1.f + sbv[b * 2 * HC + hh * CCH + c];
      float bt = sbv[b * 2 * HC + HC + hh * CCH + c];
      vloc[c] = g * xn + bt;
    }
    float4* dst = (float4*)(vals + ((size_t)bh * NPTS + n) * CCH);
#pragma unroll
    for (int q = 0; q < 8; ++q)
      dst[q] = make_float4(vloc[4 * q], vloc[4 * q + 1], vloc[4 * q + 2], vloc[4 * q + 3]);
  }
}

// ---------------------------------------------------------------- splat (scatter-add atomics)
__global__ void k_splat(const int* __restrict__ idxb, const float* __restrict__ wb,
                        const float* __restrict__ vals, float* __restrict__ grid) {
  const int bh = blockIdx.y;
  const int corner = threadIdx.x & 7;
  const int n = blockIdx.x * 32 + (threadIdx.x >> 3);
  const int idx = idxb[((size_t)bh * 8 + corner) * NPTS + n];
  const float w = wb[((size_t)bh * 8 + corner) * NPTS + n];
  const float* v = vals + ((size_t)bh * NPTS + n) * CCH;
  float* g = grid + ((size_t)bh * G3 + idx) * CCH;
#pragma unroll
  for (int c = 0; c < CCH; ++c) unsafeAtomicAdd(&g[c], v[c] * w);
}

// ---------------------------------------------------------------- grouped 3x3x3 conv (WMMA bf16)
// out[bh,v,co] = bias + sum_{tap,ci} W[h,co,ci,tap] * grid[bh, v+off(tap), ci]
// Block: 256 thr = 8 waves, 128 consecutive voxels (4 y-rows at fixed z).
// Per dz slab: stage 9-tap weights (fragment order, 18KB) + 6y x 32x neighborhood tile (12KB) in LDS.
__global__ void k_conv(const __bf16* __restrict__ gridbf, const float* __restrict__ conv_w,
                       const float* __restrict__ conv_b, float* __restrict__ out) {
  __shared__ __attribute__((aligned(32))) __bf16 lWa[9 * 2 * 32 * 16]; // [tq][tile][lane][e]
  __shared__ __attribute__((aligned(64))) __bf16 lT[6 * 32 * CCH];    // [(ry*32+x)][ci]
  const int bh = blockIdx.y;
  const int hh = bh & 7;
  const int base = blockIdx.x * 128;
  const int z = base >> 10;
  const int y0 = (base >> 5) & 31;
  const int wave = threadIdx.x >> 5, lane = threadIdx.x & 31;
  const int h = lane >> 4, lm = lane & 15;
  const int lvox = wave * 16 + lm;
  const int x = lvox & 31;
  const int yl = lvox >> 5;                     // 0..3
  const __bf16* gb = gridbf + (size_t)bh * G3 * CCH;
  const float* wsrc = conv_w + (size_t)hh * CCH * CCH * 27;
  v8f acc0 = {}, acc1 = {};
  for (int dz = -1; dz <= 1; ++dz) {
    __syncthreads();
    // stage weights for this dz slab, pre-swizzled to fragment order
    for (int i = threadIdx.x; i < 9 * 2 * 32 * 16; i += 256) {
      int e = i & 15, l = (i >> 4) & 31, tile = (i >> 9) & 1, tq = i >> 10;
      int co = tile * 16 + (l & 15);
      int ci = a_kmap(e, l >> 4);
      lWa[i] = (__bf16)wsrc[((size_t)co * CCH + ci) * 27 + (dz + 1) * 9 + tq];
    }
    // stage neighborhood tile: rows y0-1..y0+4 of plane z+dz (zero OOB)
    int zp = z + dz;
    for (int i = threadIdx.x; i < 6 * 32 * 4; i += 256) {   // 16B units
      int q = i & 3, xx = (i >> 2) & 31, ry = i >> 7;
      int yy = y0 - 1 + ry;
      uint4 v = make_uint4(0u, 0u, 0u, 0u);
      if ((unsigned)zp < 32u && (unsigned)yy < 32u)
        v = ((const uint4*)(gb + ((size_t)((zp * GG + yy) * GG + xx)) * CCH))[q];
      ((uint4*)lT)[(size_t)(ry * 32 + xx) * 4 + q] = v;
    }
    __syncthreads();
    for (int tq = 0; tq < 9; ++tq) {
      int dy = tq / 3 - 1, dx = tq % 3 - 1;
      int ry = yl + 1 + dy;                     // 0..5
      int nx = x + dx;
      v16bf bf;
      if ((unsigned)nx < 32u) {
        bf = *(const v16bf*)&lT[(size_t)(ry * 32 + nx) * CCH + 16 * h];
      } else {
        bf = {};
      }
      v16bf a0 = *(const v16bf*)&lWa[((tq * 2 + 0) * 32 + lane) * 16];
      v16bf a1 = *(const v16bf*)&lWa[((tq * 2 + 1) * 32 + lane) * 16];
      acc0 = __builtin_amdgcn_wmma_f32_16x16x32_bf16(false, a0, false, bf, (short)0, acc0, false, false);
      acc1 = __builtin_amdgcn_wmma_f32_16x16x32_bf16(false, a1, false, bf, (short)0, acc1, false, false);
    }
  }
  const int vox = base + lvox;
  float* ob = out + ((size_t)bh * G3 + vox) * CCH;
  const float* cb = conv_b + hh * CCH;
  float4 s;
  s = make_float4(acc0[0] + cb[8 * h + 0], acc0[1] + cb[8 * h + 1],
                  acc0[2] + cb[8 * h + 2], acc0[3] + cb[8 * h + 3]);
  *(float4*)(ob + 8 * h) = s;
  s = make_float4(acc0[4] + cb[8 * h + 4], acc0[5] + cb[8 * h + 5],
                  acc0[6] + cb[8 * h + 6], acc0[7] + cb[8 * h + 7]);
  *(float4*)(ob + 8 * h + 4) = s;
  s = make_float4(acc1[0] + cb[16 + 8 * h + 0], acc1[1] + cb[16 + 8 * h + 1],
                  acc1[2] + cb[16 + 8 * h + 2], acc1[3] + cb[16 + 8 * h + 3]);
  *(float4*)(ob + 16 + 8 * h) = s;
  s = make_float4(acc1[4] + cb[16 + 8 * h + 4], acc1[5] + cb[16 + 8 * h + 5],
                  acc1[6] + cb[16 + 8 * h + 6], acc1[7] + cb[16 + 8 * h + 7]);
  *(float4*)(ob + 16 + 8 * h + 4) = s;
}

// ---------------------------------------------------------------- slice (trilinear gather)
__global__ void k_slice(const float* __restrict__ conv, const int* __restrict__ idxb,
                        const float* __restrict__ wb, float* __restrict__ sliced) {
  const int b = blockIdx.y;
  const int n = blockIdx.x * 256 + threadIdx.x;
  for (int hh = 0; hh < NH; ++hh) {
    const int bh = b * NH + hh;
    float acc[CCH];
#pragma unroll
    for (int c = 0; c < CCH; ++c) acc[c] = 0.f;
    for (int o = 0; o < 8; ++o) {
      int idx = idxb[((size_t)bh * 8 + o) * NPTS + n];
      float w = wb[((size_t)bh * 8 + o) * NPTS + n];
      const float4* row = (const float4*)(conv + ((size_t)bh * G3 + idx) * CCH);
#pragma unroll
      for (int q = 0; q < 8; ++q) {
        float4 r4 = row[q];
        acc[4 * q] += w * r4.x; acc[4 * q + 1] += w * r4.y;
        acc[4 * q + 2] += w * r4.z; acc[4 * q + 3] += w * r4.w;
      }
    }
    for (int c = 0; c < CCH; ++c)
      sliced[((size_t)b * HC + hh * CCH + c) * NPTS + n] = acc[c];
  }
}

// ---------------------------------------------------------------- final adain + relu
__global__ void k_final(const float* __restrict__ sliced, const float* __restrict__ mean,
                        const float* __restrict__ istd, const float* __restrict__ sba,
                        float* __restrict__ outp) {
  const int b = blockIdx.y;
  const int n = blockIdx.x * 256 + threadIdx.x;
  for (int ch = 0; ch < HC; ++ch) {
    float xn = (sliced[((size_t)b * HC + ch) * NPTS + n] - mean[b * HC + ch]) * istd[b * HC + ch];
    float g = 1.f + sba[b * 2 * HC + ch], bt = sba[b * 2 * HC + HC + ch];
    outp[((size_t)b * HC + ch) * NPTS + n] = fmaxf(g * xn + bt, 0.f);
  }
}

extern "C" void kernel_launch(void* const* d_in, const int* in_sizes, int n_in,
                              void* d_out, int out_size, void* d_ws, size_t ws_size,
                              hipStream_t stream) {
  (void)in_sizes; (void)n_in; (void)out_size; (void)ws_size;
  const float* x      = (const float*)d_in[0];
  const float* orig   = (const float*)d_in[1];
  const float* style  = (const float*)d_in[2];
  const float* kv_w   = (const float*)d_in[3];
  const float* keys_W = (const float*)d_in[4];
  const float* keys_b = (const float*)d_in[5];
  const float* val_W  = (const float*)d_in[6];
  const float* val_b  = (const float*)d_in[7];
  const float* scale  = (const float*)d_in[8];
  const float* R      = (const float*)d_in[9];
  const float* t      = (const float*)d_in[10];
  const float* conv_w = (const float*)d_in[11];
  const float* conv_b = (const float*)d_in[12];
  const float* aft_W  = (const float*)d_in[13];
  const float* aft_b  = (const float*)d_in[14];

  float* ws = (float*)d_ws;
  size_t off = 0;
  __bf16* xT     = (__bf16*)(ws + off); off += (size_t)BN * NPTS * INCH / 2;      // bf16
  __bf16* gridbf = (__bf16*)(ws + off); off += (size_t)BN * NH * G3 * CCH / 2;    // bf16
  float* kv      = ws + off; off += (size_t)BN * KVC * NPTS;
  float* mean_kv = ws + off; off += (size_t)BN * KVC;
  float* istd_kv = ws + off; off += (size_t)BN * KVC;
  float* sbk     = ws + off; off += (size_t)BN * 48;
  float* sbv     = ws + off; off += (size_t)BN * 2 * HC;
  float* sba     = ws + off; off += (size_t)BN * 2 * HC;
  int*   idxb    = (int*)(ws + off); off += (size_t)BN * NH * 8 * NPTS;
  float* wbuf    = ws + off; off += (size_t)BN * NH * 8 * NPTS;
  float* vals    = ws + off; off += (size_t)BN * NH * NPTS * CCH;
  float* gridf   = ws + off; off += (size_t)BN * NH * G3 * CCH;
  float* mean_sl = ws + off; off += (size_t)BN * HC;
  float* istd_sl = ws + off; off += (size_t)BN * HC;
  float* convout = gridf;   // alias: gridf dead after k_cvtg
  float* sliced  = vals;    // alias: vals dead after k_splat

  const long long gridN = (long long)BN * NH * G3 * CCH;
  k_zero<<<(unsigned)((gridN + 255) / 256), 256, 0, stream>>>(gridf, gridN);

  k_xpose<<<dim3(NPTS / 64, INCH / 32, BN), 256, 0, stream>>>(x, xT);
  k_kv_gemm<<<dim3(64, 18, BN), 128, 0, stream>>>(xT, kv_w, kv);
  k_stats<<<dim3(KVC, BN), 256, 0, stream>>>(kv, KVC, mean_kv, istd_kv);

  k_affine<<<dim3(48, BN), 256, 0, stream>>>(style, keys_W, keys_b, sbk, 48);
  k_affine<<<dim3(2 * HC, BN), 256, 0, stream>>>(style, val_W, val_b, sbv, 2 * HC);
  k_affine<<<dim3(2 * HC, BN), 256, 0, stream>>>(style, aft_W, aft_b, sba, 2 * HC);

  k_prep<<<dim3(NPTS / 256, BN), 256, 0, stream>>>(kv, mean_kv, istd_kv, sbk, sbv, orig,
                                                   scale, R, t, idxb, wbuf, vals);
  k_splat<<<dim3(NPTS / 32, BN * NH), 256, 0, stream>>>(idxb, wbuf, vals, gridf);
  k_cvtg<<<(unsigned)((gridN / 2 + 255) / 256), 256, 0, stream>>>(gridf, gridbf, gridN / 2);
  k_conv<<<dim3(G3 / 128, BN * NH), 256, 0, stream>>>(gridbf, conv_w, conv_b, convout);
  k_slice<<<dim3(NPTS / 256, BN), 256, 0, stream>>>(convout, idxb, wbuf, sliced);
  k_stats<<<dim3(HC, BN), 256, 0, stream>>>(sliced, HC, mean_sl, istd_sl);
  k_final<<<dim3(NPTS / 256, BN), 256, 0, stream>>>(sliced, mean_sl, istd_sl, sba,
                                                    (float*)d_out);
}
